// GRU_83159156785365
// MI455X (gfx1250) — matrix operand order repaired and checked by reading, not compiled, not run on hardware
//
#include <hip/hip_runtime.h>
#include <hip/hip_bf16.h>

// ---------------------------------------------------------------------------
// 2-layer GRU + FC for MI455X (gfx1250, wave32, WMMA bf16 16x16x32).
//   K0: f32 -> bf16 conversion of x and all weight matrices
//   K1: time-batched input projection GI0 = X @ W_ih0^T + b_ih0  (big GEMM)
//   K2: persistent recurrent kernel, 4 workgroups x 1024 threads, batch-split
//   K3: final FC logits = H1 @ W_fc^T + b_fc                      (big GEMM)
// ---------------------------------------------------------------------------

typedef __attribute__((ext_vector_type(16))) __bf16 v16bf;
typedef __attribute__((ext_vector_type(8)))  float  v8f;

#define GRU_B 64
#define GRU_S 1024
#define GRU_I 256
#define GRU_H 512
#define GRU_O 512
#define GRU_G 1536   // 3*H

__device__ __forceinline__ float sigmoidf_(float x) {
    return 1.0f / (1.0f + __expf(-x));
}

// A fragment: row-major [rows, K] bf16, tile at (row0, k0).
// ISA layout (16-bit A 16x32): lanes 0-15 -> M=lane, K = k0+{0..7, 16..23};
//                              lanes 16-31 -> M=lane-16, K = k0+{8..15, 24..31}.
__device__ __forceinline__ v16bf load_a_frag(const __bf16* A, int lda,
                                             int row0, int k0, int lane) {
    const int half = lane >> 4;
    const int r    = lane & 15;
    const __bf16* p = A + (size_t)(row0 + r) * lda + k0 + half * 8;
    v16bf a;
#pragma unroll
    for (int i = 0; i < 8; ++i) {
        a[i]     = p[i];
        a[8 + i] = p[16 + i];
    }
    return a;
}

// B fragment: weights W row-major [N, K] (used as B = W^T, K x N), tile at
// (n0, k0). Layout (16-bit B 32x16), matching A: lanes 0-15 -> N=lane, K=k0+{0..7, 16..23};
//                                        lanes 16-31 -> N=lane-16, K=k0+{8..15, 24..31}.
// Both are two 8-element runs of row (n0+r) of W.
__device__ __forceinline__ v16bf load_b_frag(const __bf16* W, int ldw,
                                             int n0, int k0, int lane) {
    const int half = lane >> 4;
    const int r    = lane & 15;
    const __bf16* p = W + (size_t)(n0 + r) * ldw + k0 + half * 8;
    v16bf b;
#pragma unroll
    for (int i = 0; i < 8; ++i) {
        b[i]     = p[i];
        b[8 + i] = p[16 + i];
    }
    return b;
}

// ---------------------------------------------------------------------------
__global__ void f32_to_bf16_kernel(const float* __restrict__ in,
                                   __bf16* __restrict__ out, int n) {
    int i = blockIdx.x * blockDim.x + threadIdx.x;
    if (i < n) out[i] = (__bf16)in[i];
}

// ---------------------------------------------------------------------------
// Generic bf16 GEMM: out[rows, cols] = A[rows,K] @ W[cols,K]^T + bias.
// One 16x16 output tile per wave; 8 waves per 256-thread block.
__global__ void gemm_bf16_kernel(const __bf16* __restrict__ A,
                                 const __bf16* __restrict__ W,
                                 const float* __restrict__ bias,
                                 void* __restrict__ out, int out_is_bf16,
                                 int rows, int cols, int K) {
    const int lane = threadIdx.x & 31;
    const int wave = blockIdx.x * (blockDim.x >> 5) + (threadIdx.x >> 5);
    const int tiles_n = cols >> 4;
    const int tm = wave / tiles_n;
    const int tn = wave - tm * tiles_n;
    if (tm * 16 >= rows) return;              // wave-uniform guard
    const int row0 = tm * 16, n0 = tn * 16;
    const int half = lane >> 4, r = lane & 15;

    const float bv = bias[n0 + r];
    v8f c;
#pragma unroll
    for (int j = 0; j < 8; ++j) c[j] = bv;

    for (int k0 = 0; k0 < K; k0 += 32) {
        v16bf a = load_a_frag(A, K, row0, k0, lane);
        v16bf b = load_b_frag(W, K, n0, k0, lane);
        c = __builtin_amdgcn_wmma_f32_16x16x32_bf16(false, a, false, b,
                                                    (short)0, c, false, false);
    }

    // D layout: VGPR j -> M = j + 8*half, N = n0 + r
    if (out_is_bf16) {
        __bf16* o = (__bf16*)out;
#pragma unroll
        for (int j = 0; j < 8; ++j)
            o[(size_t)(row0 + half * 8 + j) * cols + n0 + r] = (__bf16)c[j];
    } else {
        float* o = (float*)out;
#pragma unroll
        for (int j = 0; j < 8; ++j)
            o[(size_t)(row0 + half * 8 + j) * cols + n0 + r] = c[j];
    }
}

// ---------------------------------------------------------------------------
// Recurrent kernel. 4 blocks x 1024 threads (32 waves). Block owns 16 batch
// rows; wave w owns hidden columns [16w, 16w+16). Hidden state in LDS (bf16).
__global__ __launch_bounds__(1024)
void gru_recurrent_kernel(const __bf16* __restrict__ GI0,   // [B*S, 3H]
                          const __bf16* __restrict__ Whh0,  // [3H, H]
                          const float*  __restrict__ bhh0,
                          const __bf16* __restrict__ Wih1,  // [3H, H]
                          const float*  __restrict__ bih1,
                          const __bf16* __restrict__ Whh1,  // [3H, H]
                          const float*  __restrict__ bhh1,
                          __bf16* __restrict__ H1,          // [B*S, H]
                          float*  __restrict__ hid) {       // [2, B, H]
    __shared__ __bf16 h0s[16][GRU_H];
    __shared__ __bf16 h1s[16][GRU_H];

    const int tid  = threadIdx.x;
    const int lane = tid & 31;
    const int w    = tid >> 5;       // wave 0..31 == hidden column group
    const int half = lane >> 4;
    const int r    = lane & 15;
    const int m0   = blockIdx.x * 16;   // batch rows of this block
    const int n0   = w * 16;
    const int col  = n0 + r;

    for (int i = tid; i < 16 * GRU_H; i += blockDim.x) {
        (&h0s[0][0])[i] = (__bf16)0.0f;
        (&h1s[0][0])[i] = (__bf16)0.0f;
    }
    __syncthreads();

    const float b0r = bhh0[col], b0z = bhh0[GRU_H + col], b0n = bhh0[2 * GRU_H + col];
    const float i1r = bih1[col], i1z = bih1[GRU_H + col], i1n = bih1[2 * GRU_H + col];
    const float b1r = bhh1[col], b1z = bhh1[GRU_H + col], b1n = bhh1[2 * GRU_H + col];

    for (int s = 0; s < GRU_S; ++s) {
        // ---- Layer 0: gh0 = h0 @ Whh0^T + b_hh0, gates for cols [n0,n0+16)
        v8f cr, cz, cn;
#pragma unroll
        for (int j = 0; j < 8; ++j) { cr[j] = b0r; cz[j] = b0z; cn[j] = b0n; }
        for (int k0 = 0; k0 < GRU_H; k0 += 32) {
            v16bf a  = load_a_frag(&h0s[0][0], GRU_H, 0, k0, lane);
            v16bf br = load_b_frag(Whh0, GRU_H, 0 * GRU_H + n0, k0, lane);
            v16bf bz = load_b_frag(Whh0, GRU_H, 1 * GRU_H + n0, k0, lane);
            v16bf bn = load_b_frag(Whh0, GRU_H, 2 * GRU_H + n0, k0, lane);
            cr = __builtin_amdgcn_wmma_f32_16x16x32_bf16(false, a, false, br, (short)0, cr, false, false);
            cz = __builtin_amdgcn_wmma_f32_16x16x32_bf16(false, a, false, bz, (short)0, cz, false, false);
            cn = __builtin_amdgcn_wmma_f32_16x16x32_bf16(false, a, false, bn, (short)0, cn, false, false);
        }
        float h0new[8];
#pragma unroll
        for (int j = 0; j < 8; ++j) {
            const int m = half * 8 + j;
            const __bf16* gp = GI0 + ((size_t)(m0 + m) * GRU_S + s) * GRU_G;
            const float gr = (float)gp[col];
            const float gz = (float)gp[GRU_H + col];
            const float gn = (float)gp[2 * GRU_H + col];
            const float rr = sigmoidf_(gr + cr[j]);
            const float zz = sigmoidf_(gz + cz[j]);
            const float nn = tanhf(gn + rr * cn[j]);
            const float ho = (float)h0s[m][col];
            h0new[j] = (1.0f - zz) * nn + zz * ho;
        }
        __syncthreads();   // all waves done reading h0s
#pragma unroll
        for (int j = 0; j < 8; ++j) h0s[half * 8 + j][col] = (__bf16)h0new[j];
        __syncthreads();   // h0s fully updated

        // ---- Layer 1: gi1 = h0n @ Wih1^T + b_ih1 ; gh1 = h1 @ Whh1^T + b_hh1
        v8f er, ez, en, dr, dz, dn;
#pragma unroll
        for (int j = 0; j < 8; ++j) {
            er[j] = i1r; ez[j] = i1z; en[j] = i1n;
            dr[j] = b1r; dz[j] = b1z; dn[j] = b1n;
        }
        for (int k0 = 0; k0 < GRU_H; k0 += 32) {
            v16bf a0 = load_a_frag(&h0s[0][0], GRU_H, 0, k0, lane);
            v16bf a1 = load_a_frag(&h1s[0][0], GRU_H, 0, k0, lane);
            v16bf wr = load_b_frag(Wih1, GRU_H, 0 * GRU_H + n0, k0, lane);
            v16bf wz = load_b_frag(Wih1, GRU_H, 1 * GRU_H + n0, k0, lane);
            v16bf wn = load_b_frag(Wih1, GRU_H, 2 * GRU_H + n0, k0, lane);
            v16bf ur = load_b_frag(Whh1, GRU_H, 0 * GRU_H + n0, k0, lane);
            v16bf uz = load_b_frag(Whh1, GRU_H, 1 * GRU_H + n0, k0, lane);
            v16bf un = load_b_frag(Whh1, GRU_H, 2 * GRU_H + n0, k0, lane);
            er = __builtin_amdgcn_wmma_f32_16x16x32_bf16(false, a0, false, wr, (short)0, er, false, false);
            ez = __builtin_amdgcn_wmma_f32_16x16x32_bf16(false, a0, false, wz, (short)0, ez, false, false);
            en = __builtin_amdgcn_wmma_f32_16x16x32_bf16(false, a0, false, wn, (short)0, en, false, false);
            dr = __builtin_amdgcn_wmma_f32_16x16x32_bf16(false, a1, false, ur, (short)0, dr, false, false);
            dz = __builtin_amdgcn_wmma_f32_16x16x32_bf16(false, a1, false, uz, (short)0, dz, false, false);
            dn = __builtin_amdgcn_wmma_f32_16x16x32_bf16(false, a1, false, un, (short)0, dn, false, false);
        }
        float h1new[8];
#pragma unroll
        for (int j = 0; j < 8; ++j) {
            const int m = half * 8 + j;
            const float rr = sigmoidf_(er[j] + dr[j]);
            const float zz = sigmoidf_(ez[j] + dz[j]);
            const float nn = tanhf(en[j] + rr * dn[j]);
            const float ho = (float)h1s[m][col];
            h1new[j] = (1.0f - zz) * nn + zz * ho;
        }
        __syncthreads();   // all waves done reading h1s
#pragma unroll
        for (int j = 0; j < 8; ++j) {
            const int m = half * 8 + j;
            h1s[m][col] = (__bf16)h1new[j];
            H1[((size_t)(m0 + m) * GRU_S + s) * GRU_H + col] = (__bf16)h1new[j];
        }
        __syncthreads();   // h1s fully updated
    }

    // Final hidden states -> d_out tail: hid[0][b][h], hid[1][b][h] (f32)
#pragma unroll
    for (int j = 0; j < 8; ++j) {
        const int m = half * 8 + j;
        hid[(size_t)(m0 + m) * GRU_H + col] = (float)h0s[m][col];
        hid[(size_t)GRU_B * GRU_H + (size_t)(m0 + m) * GRU_H + col] = (float)h1s[m][col];
    }
}

// ---------------------------------------------------------------------------
extern "C" void kernel_launch(void* const* d_in, const int* in_sizes, int n_in,
                              void* d_out, int out_size, void* d_ws, size_t ws_size,
                              hipStream_t stream) {
    const float* x    = (const float*)d_in[0];
    const float* Wih0 = (const float*)d_in[1];
    const float* Whh0 = (const float*)d_in[2];
    const float* bih0 = (const float*)d_in[3];
    const float* bhh0 = (const float*)d_in[4];
    const float* Wih1 = (const float*)d_in[5];
    const float* Whh1 = (const float*)d_in[6];
    const float* bih1 = (const float*)d_in[7];
    const float* bhh1 = (const float*)d_in[8];
    const float* Wfc  = (const float*)d_in[9];
    const float* bfc  = (const float*)d_in[10];

    char* ws = (char*)d_ws;
    size_t off = 0;
    auto alloc = [&](size_t bytes) -> char* {
        char* p = ws + off;
        off += (bytes + 255) & ~(size_t)255;
        return p;
    };

    const int nx    = GRU_B * GRU_S * GRU_I;           // 16,777,216
    const int nWih0 = GRU_G * GRU_I;
    const int nWhh0 = GRU_G * GRU_H;
    const int nWih1 = GRU_G * GRU_H;
    const int nWhh1 = GRU_G * GRU_H;
    const int nWfc  = GRU_O * GRU_H;
    const size_t rowsBS = (size_t)GRU_B * GRU_S;       // 65,536

    __bf16* xbf     = (__bf16*)alloc((size_t)nx * 2);
    __bf16* GI0bf   = (__bf16*)alloc(rowsBS * GRU_G * 2);   // 192 MiB
    __bf16* H1bf    = (__bf16*)alloc(rowsBS * GRU_H * 2);   //  64 MiB
    __bf16* Wih0bf  = (__bf16*)alloc((size_t)nWih0 * 2);
    __bf16* Whh0bf  = (__bf16*)alloc((size_t)nWhh0 * 2);
    __bf16* Wih1bf  = (__bf16*)alloc((size_t)nWih1 * 2);
    __bf16* Whh1bf  = (__bf16*)alloc((size_t)nWhh1 * 2);
    __bf16* Wfcbf   = (__bf16*)alloc((size_t)nWfc * 2);

    // K0: conversions
    auto cvt = [&](const float* src, __bf16* dst, int n) {
        f32_to_bf16_kernel<<<(n + 255) / 256, 256, 0, stream>>>(src, dst, n);
    };
    cvt(x, xbf, nx);
    cvt(Wih0, Wih0bf, nWih0);
    cvt(Whh0, Whh0bf, nWhh0);
    cvt(Wih1, Wih1bf, nWih1);
    cvt(Whh1, Whh1bf, nWhh1);
    cvt(Wfc,  Wfcbf,  nWfc);

    // K1: GI0 = X @ Wih0^T + b_ih0   (65536 x 1536, K=256) -> bf16
    {
        const int tiles = (int)(rowsBS / 16) * (GRU_G / 16);   // 393216
        gemm_bf16_kernel<<<tiles / 8, 256, 0, stream>>>(
            xbf, Wih0bf, bih0, (void*)GI0bf, /*out_is_bf16=*/1,
            (int)rowsBS, GRU_G, GRU_I);
    }

    // K2: recurrence (4 workgroups x 1024 threads, 16 batch rows each)
    float* logits = (float*)d_out;
    float* hid    = logits + rowsBS * GRU_O;
    gru_recurrent_kernel<<<GRU_B / 16, 1024, 0, stream>>>(
        GI0bf, Whh0bf, bhh0, Wih1bf, bih1, Whh1bf, bhh1, H1bf, hid);

    // K3: logits = H1 @ Wfc^T + b_fc  (65536 x 512, K=512) -> f32
    {
        const int tiles = (int)(rowsBS / 16) * (GRU_O / 16);   // 131072
        gemm_bf16_kernel<<<tiles / 8, 256, 0, stream>>>(
            H1bf, Wfcbf, bfc, (void*)logits, /*out_is_bf16=*/0,
            (int)rowsBS, GRU_O, GRU_H);
    }
}
